// TransformerBlock_47055661695477
// MI455X (gfx1250) — compile-verified, matrix operand order checked by reading, not run on hardware
//
#include <hip/hip_runtime.h>
#include <math.h>

// ---------------------------------------------------------------------------
// GPT-2 style transformer block on gfx1250 (CDNA5), bf16 WMMA everywhere.
// D_EMB=768, D_MLP=3072, N_HEAD=12, D_HEAD=64, B=2, T=4096 -> 8192 rows.
// ---------------------------------------------------------------------------

typedef unsigned short u16;
typedef __attribute__((ext_vector_type(16))) __bf16 v16bf;
typedef __attribute__((ext_vector_type(8)))  float  v8f;

// TDM descriptor vector types (6-arg clang-23 form:
//   (uint32x4 g0, int32x8 g1, int32x4 g2, int32x4 g3, int32x8 g4, i32 cpol))
typedef __attribute__((ext_vector_type(4))) unsigned int u32x4;
typedef __attribute__((ext_vector_type(8))) int          i32x8;
typedef __attribute__((ext_vector_type(4))) int          i32x4;

#if defined(__HIP_DEVICE_COMPILE__) && defined(__gfx1250__) && \
    __has_builtin(__builtin_amdgcn_tensor_load_to_lds) &&       \
    __has_builtin(__builtin_amdgcn_s_wait_tensorcnt)
#define HAS_TDM 1
#else
#define HAS_TDM 0
#endif

union AFrag {          // one 16x32 bf16 WMMA operand (A or B^T view): 32 bytes/lane
  v16bf v;
  uint4 q[2];
};

__device__ __forceinline__ u16 f32_bf16(float f) {
  unsigned u = __float_as_uint(f);
  u += 0x7FFFu + ((u >> 16) & 1u);       // round-to-nearest-even
  return (u16)(u >> 16);
}

// ---------------------------------------------------------------------------
// fp32 -> bf16 elementwise convert
// ---------------------------------------------------------------------------
__global__ __launch_bounds__(256) void k_cvt_bf16(const float* __restrict__ in,
                                                  u16* __restrict__ out, int n) {
  int i = blockIdx.x * 256 + threadIdx.x;
  if (i < n) out[i] = f32_bf16(in[i]);
}

// ---------------------------------------------------------------------------
// LayerNorm: one wave (32 lanes) per 768-wide row, 8 rows per 256-thread block
// ---------------------------------------------------------------------------
__global__ __launch_bounds__(256) void k_layernorm(const float* __restrict__ x,
                                                   const float* __restrict__ g,
                                                   const float* __restrict__ b,
                                                   u16* __restrict__ out) {
  const int lane = threadIdx.x & 31;
  const int wave = threadIdx.x >> 5;
  const int row  = blockIdx.x * 8 + wave;
  const float* xr = x + (size_t)row * 768;

  float vals[24], s = 0.f, s2 = 0.f;
#pragma unroll
  for (int i = 0; i < 24; ++i) {
    float v = xr[lane + i * 32];
    vals[i] = v; s += v; s2 += v * v;
  }
#pragma unroll
  for (int off = 16; off >= 1; off >>= 1) {
    s  += __shfl_xor(s,  off, 32);
    s2 += __shfl_xor(s2, off, 32);
  }
  const float mean = s * (1.f / 768.f);
  const float var  = s2 * (1.f / 768.f) - mean * mean;
  const float rstd = rsqrtf(var + 1e-5f);

  u16* orow = out + (size_t)row * 768;
#pragma unroll
  for (int i = 0; i < 24; ++i) {
    const int c = lane + i * 32;
    orow[c] = f32_bf16((vals[i] - mean) * rstd * g[c] + b[c]);
  }
}

// ---------------------------------------------------------------------------
// bf16 GEMM: C[M,N] = A[M,K] * B[K,N]  (+bias, +residual, optional GELU)
// Block tile 128x128, K-step 32, 8 waves; wave tile 32x64 = 2x4 WMMA accums.
// Fully templated so every call site specializes (no runtime branches, the
// N/K strides fold into immediates). A tile staged into LDS by the Tensor
// Data Mover (TDM) with its pad feature reproducing the LDA=40 layout;
// B tile staged transposed by the threads.
// ---------------------------------------------------------------------------
#define G_LDA 40   // padded LDS row stride (ushorts); 80B keeps 16B alignment
#define G_LDB 40

template <int N, int K, bool GELU, bool OUTBF, bool BIAS, bool RES>
__global__ __launch_bounds__(256)
void k_gemm(const u16* __restrict__ A, const u16* __restrict__ B,
            const float* __restrict__ bias, const float* __restrict__ resid,
            void* __restrict__ out) {
  __shared__ u16 As[128 * G_LDA];
  __shared__ u16 Bt[128 * G_LDB];   // B tile stored transposed: [n][k]

  const int t    = threadIdx.x;
  const int lane = t & 31;
  const int wave = t >> 5;
  const int wm   = wave >> 1;               // 0..3 (M direction, 32 rows each)
  const int wn   = wave & 1;                // 0..1 (N direction, 64 cols each)
  const int bm   = blockIdx.y * 128;
  const int bn   = blockIdx.x * 128;

  const int lrow  = lane & 15;
  const int lo    = (lane & 16) ? 8 : 0;    // per-lane K chunk offset (ISA layout)
  const int half8 = (lane & 16) ? 8 : 0;

  v8f acc[2][4];
  const v8f vzero = {0.f, 0.f, 0.f, 0.f, 0.f, 0.f, 0.f, 0.f};
#pragma unroll
  for (int sm = 0; sm < 2; ++sm)
#pragma unroll
    for (int sn = 0; sn < 4; ++sn) acc[sm][sn] = vzero;

  const int arow = t >> 1, aseg = (t & 1) * 16;       // A loader mapping (fallback)
  const int bkr  = t >> 3, bseg = (t & 7) * 16;       // B loader mapping

#if HAS_TDM
  // --- TDM descriptor (group0/group1), 2D tile 32(K) x 128(M) of 2B elems ---
  const unsigned lds_as =
      (unsigned)(unsigned long long)(__attribute__((address_space(3))) char*)As;
  const unsigned long long ga_base =
      (unsigned long long)(const void*)(A + (size_t)bm * K);
  i32x8 g1;
  // [17:16] data_size=1 (2B); [20] pad_enable; [24:22] pad_interval=3 (16 DW
  // = one 64B row); [31:25] pad_amount=3 (4 DW = 8 u16 -> stride 40)
  g1[0] = (int)((1u << 16) | (1u << 20) | (3u << 22) | (3u << 25));
  g1[1] = (int)((unsigned)(K & 0xFFFF) << 16);          // tensor_dim0 = K (lo16)
  g1[2] = (int)(((unsigned)(K >> 16) & 0xFFFF) | (8192u << 16)); // dim0 hi / dim1 lo
  g1[3] = (int)(32u << 16);                             // tensor_dim1 hi=0, tile_dim0=32
  g1[4] = (int)128;                                     // tile_dim1=128, tile_dim2=0
  g1[5] = (int)(unsigned)K;                             // tensor_dim0_stride = K
  g1[6] = 0;                                            // stride hi / dim1_stride lo
  g1[7] = 0;
  const i32x4 gz4 = {0, 0, 0, 0};
  const i32x8 gz8 = {0, 0, 0, 0, 0, 0, 0, 0};
#endif

  for (int k0 = 0; k0 < K; k0 += 32) {
    __syncthreads();
#if HAS_TDM
    if (wave == 0) {   // one TDM DMA per block per K-step (EXEC-independent)
      const unsigned long long ga = ga_base + (unsigned)k0 * 2u;
      u32x4 g0;
      g0[0] = 1u;                                        // count=1, user mode
      g0[1] = lds_as;                                    // LDS byte address
      g0[2] = (unsigned)(ga & 0xFFFFFFFFu);
      g0[3] = (unsigned)((ga >> 32) & 0x1FFFFFFu) | (2u << 30);  // addr[56:32], type=2
      __builtin_amdgcn_tensor_load_to_lds(g0, g1, gz4, gz4, gz8, 0);
    }
#else
    {
      const uint4* ag = (const uint4*)(A + (size_t)(bm + arow) * K + k0 + aseg);
      *(uint4*)(&As[arow * G_LDA + aseg])     = ag[0];
      *(uint4*)(&As[arow * G_LDA + aseg + 8]) = ag[1];
    }
#endif
    // stage B tile (32x128) transposed -> Bt[n][k]  (overlaps the TDM DMA)
    {
      union { uint4 q[2]; u16 u[16]; } bv;
      const u16* bg = B + (size_t)(k0 + bkr) * N + bn + bseg;
      bv.q[0] = ((const uint4*)bg)[0];
      bv.q[1] = ((const uint4*)bg)[1];
      if (k0 + 32 < K) __builtin_prefetch(bg + (size_t)32 * N, 0, 1);
#pragma unroll
      for (int e = 0; e < 16; ++e) Bt[(bseg + e) * G_LDB + bkr] = bv.u[e];
    }
#if HAS_TDM
    if (wave == 0) __builtin_amdgcn_s_wait_tensorcnt(0);
#endif
    __syncthreads();

    AFrag af[2];
#pragma unroll
    for (int sm = 0; sm < 2; ++sm) {
      const u16* p = &As[(wm * 32 + sm * 16 + lrow) * G_LDA];
      af[sm].q[0] = *(const uint4*)(p + lo);
      af[sm].q[1] = *(const uint4*)(p + lo + 16);
    }
#pragma unroll
    for (int sn = 0; sn < 4; ++sn) {
      AFrag bf;
      const u16* p = &Bt[(wn * 64 + sn * 16 + lrow) * G_LDB];
      bf.q[0] = *(const uint4*)(p + lo);
      bf.q[1] = *(const uint4*)(p + lo + 16);
#pragma unroll
      for (int sm = 0; sm < 2; ++sm) {
        acc[sm][sn] = __builtin_amdgcn_wmma_f32_16x16x32_bf16(
            false, af[sm].v, false, bf.v, (short)0, acc[sm][sn], false, false);
      }
    }
  }

  // ---- epilogue: fully specialized, no runtime branches ----
#pragma unroll
  for (int sn = 0; sn < 4; ++sn) {
    const int col = bn + wn * 64 + sn * 16 + lrow;
    float bcol = 0.f;
    if constexpr (BIAS) bcol = bias[col];
#pragma unroll
    for (int sm = 0; sm < 2; ++sm) {
      const int rbase = bm + wm * 32 + sm * 16 + half8;
#pragma unroll
      for (int i = 0; i < 8; ++i) {
        const size_t idx = (size_t)(rbase + i) * N + col;
        float v = acc[sm][sn][i] + bcol;
        if constexpr (RES) v += resid[idx];
        if constexpr (GELU) v = 0.5f * v * (1.f + erff(v * 0.70710678118f));
        if constexpr (OUTBF) ((u16*)out)[idx] = f32_bf16(v);
        else                 ((float*)out)[idx] = v;
      }
    }
  }
}

// ---------------------------------------------------------------------------
// Flash attention, causal, scale 1/8. Grid (T/64, B*H); 128 threads = 4 waves,
// each wave owns 16 query rows. QK^T and P*V both via bf16 WMMA.
// K tile staged once per block into LDS by TDM (rows already in B^T layout);
// V tile transposed into LDS by the threads.
// qkv: bf16 [B*T, 2304] = [Q|K|V] with head h at col h*64 (+0/+768/+1536).
// out: bf16 [B*T, 768]
// ---------------------------------------------------------------------------
#define A_LDK 72   // padded row stride for K tile (64 + 8 u16; 144B, 16B-aligned)

__global__ __launch_bounds__(128)
void k_attention(const u16* __restrict__ qkv, u16* __restrict__ out) {
  __shared__ u16 Ks[32 * A_LDK];     // K tile [key][d], natural (B^T) layout
  __shared__ u16 Vt[64 * 40];        // V^T tile [d][key] for current key block
  __shared__ u16 Ps[4][16 * 40];     // per-wave P staging (16x32 bf16)

  const int t    = threadIdx.x;
  const int lane = t & 31;
  const int wave = t >> 5;
  const int qb   = blockIdx.x;       // 0..63 (64-row q block)
  const int bh   = blockIdx.y;       // 0..23
  const int b    = bh / 12, h = bh % 12;
  const int q0   = qb * 64 + wave * 16;

  const int lrow  = lane & 15;
  const int lo    = (lane & 16) ? 8 : 0;
  const int half8 = (lane & 16) ? 8 : 0;
  const size_t base = (size_t)b * 4096;

  // Q fragments: 16 rows x 64 d = 2 A-fragments (K=32 each)
  AFrag qf[2];
  {
    const u16* qr = qkv + (base + q0 + lrow) * 2304 + h * 64;
#pragma unroll
    for (int f = 0; f < 2; ++f) {
      qf[f].q[0] = *(const uint4*)(qr + f * 32 + lo);
      qf[f].q[1] = *(const uint4*)(qr + f * 32 + lo + 16);
    }
  }

#if HAS_TDM
  // --- TDM descriptor for the K tile: 2D 64(d) x 32(keys), 2B elems,
  // pad 4 DW after each 32-DW (128B) row -> LDS row stride 72 u16 ---
  const unsigned lds_ks =
      (unsigned)(unsigned long long)(__attribute__((address_space(3))) char*)Ks;
  i32x8 kg1;
  kg1[0] = (int)((1u << 16) | (1u << 20) | (4u << 22) | (3u << 25));
  kg1[1] = (int)((2304u & 0xFFFF) << 16);   // tensor_dim0 = 2304 (lo16)
  kg1[2] = (int)(8192u << 16);              // tensor_dim0 hi=0, tensor_dim1 lo=8192
  kg1[3] = (int)(64u << 16);                // tensor_dim1 hi=0, tile_dim0 = 64
  kg1[4] = (int)32;                         // tile_dim1 = 32, tile_dim2 = 0
  kg1[5] = (int)2304;                       // tensor_dim0_stride = 2304
  kg1[6] = 0;
  kg1[7] = 0;
  const i32x4 gz4 = {0, 0, 0, 0};
  const i32x8 gz8 = {0, 0, 0, 0, 0, 0, 0, 0};
#endif

  const v8f vzero = {0.f, 0.f, 0.f, 0.f, 0.f, 0.f, 0.f, 0.f};
  v8f O[4];
#pragma unroll
  for (int j = 0; j < 4; ++j) O[j] = vzero;
  float mrow[8], lsum[8];
#pragma unroll
  for (int i = 0; i < 8; ++i) { mrow[i] = -1e30f; lsum[i] = 0.f; }

  const int nkb = 2 * qb + 2;                       // key blocks of 32 (causal)
  for (int kb = 0; kb < nkb; ++kb) {
    __syncthreads();
#if HAS_TDM
    if (wave == 0) {   // DMA the K tile (rows kb*32..+31, cols h*64..+63)
      const unsigned long long ga = (unsigned long long)(const void*)
          (qkv + (base + kb * 32) * 2304 + 768 + h * 64);
      u32x4 g0;
      g0[0] = 1u;
      g0[1] = lds_ks;
      g0[2] = (unsigned)(ga & 0xFFFFFFFFu);
      g0[3] = (unsigned)((ga >> 32) & 0x1FFFFFFu) | (2u << 30);
      __builtin_amdgcn_tensor_load_to_lds(g0, kg1, gz4, gz4, gz8, 0);
    }
#else
    {
      const int r = t >> 2, d0 = (t & 3) * 16;
      const u16* kr = qkv + (base + kb * 32 + r) * 2304 + 768 + h * 64 + d0;
      *(uint4*)(&Ks[r * A_LDK + d0])     = ((const uint4*)kr)[0];
      *(uint4*)(&Ks[r * A_LDK + d0 + 8]) = ((const uint4*)kr)[1];
    }
#endif
    // stage V^T cooperatively (overlaps the TDM DMA): keys kb*32..+31, d 0..63
    {
      const int r = t >> 2, d0 = (t & 3) * 16;
      union { uint4 q[2]; u16 u[16]; } vv;
      const u16* vr = qkv + (base + kb * 32 + r) * 2304 + 1536 + h * 64 + d0;
      vv.q[0] = ((const uint4*)vr)[0];
      vv.q[1] = ((const uint4*)vr)[1];
      if (kb + 1 < nkb) __builtin_prefetch(vr + 32 * 2304, 0, 1);
#pragma unroll
      for (int e = 0; e < 16; ++e) Vt[(d0 + e) * 40 + r] = vv.u[e];
    }
#if HAS_TDM
    if (wave == 0) __builtin_amdgcn_s_wait_tensorcnt(0);
#endif
    __syncthreads();

    // S = Q K^T  (16 q x 32 keys, contracted over d=64 in two steps)
    v8f S[2] = {vzero, vzero};
#pragma unroll
    for (int j = 0; j < 2; ++j) {
      const u16* kp = &Ks[(j * 16 + lrow) * A_LDK];
#pragma unroll
      for (int ds = 0; ds < 2; ++ds) {
        AFrag kf;
        kf.q[0] = *(const uint4*)(kp + ds * 32 + lo);
        kf.q[1] = *(const uint4*)(kp + ds * 32 + lo + 16);
        S[j] = __builtin_amdgcn_wmma_f32_16x16x32_bf16(
            false, qf[ds].v, false, kf.v, (short)0, S[j], false, false);
      }
    }

    // online softmax update; C-layout: VGPR i <-> row (i or i+8 per lane-half)
#pragma unroll
    for (int i = 0; i < 8; ++i) {
      const int qr   = q0 + i + half8;
      const int key0 = kb * 32 + lrow;
      float s0 = S[0][i] * 0.125f;
      float s1 = S[1][i] * 0.125f;
      if (key0 > qr)      s0 = -1e30f;
      if (key0 + 16 > qr) s1 = -1e30f;
      float mx = fmaxf(s0, s1);
#pragma unroll
      for (int off = 8; off >= 1; off >>= 1)
        mx = fmaxf(mx, __shfl_xor(mx, off, 16));
      const float mnew  = fmaxf(mrow[i], mx);
      const float alpha = __expf(mrow[i] - mnew);
      const float p0 = __expf(s0 - mnew);
      const float p1 = __expf(s1 - mnew);
      float sum = p0 + p1;
#pragma unroll
      for (int off = 8; off >= 1; off >>= 1)
        sum += __shfl_xor(sum, off, 16);
      lsum[i] = lsum[i] * alpha + sum;
      mrow[i] = mnew;
#pragma unroll
      for (int j2 = 0; j2 < 4; ++j2) O[j2][i] *= alpha;
      Ps[wave][(i + half8) * 40 + lrow]      = f32_bf16(p0);
      Ps[wave][(i + half8) * 40 + lrow + 16] = f32_bf16(p1);
    }

    // O += P * V   (same-wave LDS round trip to reach A-fragment layout)
    AFrag pf;
    {
      const u16* pp = &Ps[wave][lrow * 40];
      pf.q[0] = *(const uint4*)(pp + lo);
      pf.q[1] = *(const uint4*)(pp + lo + 16);
    }
#pragma unroll
    for (int j2 = 0; j2 < 4; ++j2) {
      AFrag vf;
      const u16* vp = &Vt[(j2 * 16 + lrow) * 40];
      vf.q[0] = *(const uint4*)(vp + lo);
      vf.q[1] = *(const uint4*)(vp + lo + 16);
      O[j2] = __builtin_amdgcn_wmma_f32_16x16x32_bf16(
          false, pf.v, false, vf.v, (short)0, O[j2], false, false);
    }
  }

  // normalize and store (bf16) into [B*T, 768] at head column
#pragma unroll
  for (int i = 0; i < 8; ++i) {
    const float inv = 1.f / lsum[i];
    const int row = q0 + i + half8;
    u16* orow = out + (base + row) * 768 + h * 64;
#pragma unroll
    for (int j2 = 0; j2 < 4; ++j2)
      orow[j2 * 16 + lrow] = f32_bf16(O[j2][i] * inv);
  }
}

// ---------------------------------------------------------------------------
// launch
// ---------------------------------------------------------------------------
extern "C" void kernel_launch(void* const* d_in, const int* in_sizes, int n_in,
                              void* d_out, int out_size, void* d_ws, size_t ws_size,
                              hipStream_t stream) {
  (void)in_sizes; (void)n_in; (void)out_size; (void)ws_size;

  const float* x      = (const float*)d_in[0];
  const float* ln1_g  = (const float*)d_in[1];
  const float* ln1_b  = (const float*)d_in[2];
  const float* wqkv_w = (const float*)d_in[3];
  const float* wo_w   = (const float*)d_in[4];
  const float* wo_b   = (const float*)d_in[5];
  const float* ln2_g  = (const float*)d_in[6];
  const float* ln2_b  = (const float*)d_in[7];
  const float* w1_w   = (const float*)d_in[8];
  const float* w1_b   = (const float*)d_in[9];
  const float* w2_w   = (const float*)d_in[10];
  const float* w2_b   = (const float*)d_in[11];

  const int ROWS = 2 * 4096;   // 8192
  char* w = (char*)d_ws;
  size_t off = 0;
  auto alloc = [&](size_t bytes) -> char* {
    char* p = w + off;
    off = (off + bytes + 255) & ~(size_t)255;
    return p;
  };
  u16*   b_wqkv = (u16*)alloc((size_t)768 * 2304 * 2);
  u16*   b_wo   = (u16*)alloc((size_t)768 * 768 * 2);
  u16*   b_w1   = (u16*)alloc((size_t)768 * 3072 * 2);
  u16*   b_w2   = (u16*)alloc((size_t)3072 * 768 * 2);
  u16*   b_ln1  = (u16*)alloc((size_t)ROWS * 768 * 2);
  u16*   b_qkv  = (u16*)alloc((size_t)ROWS * 2304 * 2);
  u16*   b_attn = (u16*)alloc((size_t)ROWS * 768 * 2);
  float* b_x1   = (float*)alloc((size_t)ROWS * 768 * 4);
  u16*   b_ln2  = (u16*)alloc((size_t)ROWS * 768 * 2);
  u16*   b_h1   = (u16*)alloc((size_t)ROWS * 3072 * 2);
  // total ~166 MB of workspace

  auto cvt = [&](const float* src, u16* dst, int n) {
    k_cvt_bf16<<<(n + 255) / 256, 256, 0, stream>>>(src, dst, n);
  };
  cvt(wqkv_w, b_wqkv, 768 * 2304);
  cvt(wo_w,   b_wo,   768 * 768);
  cvt(w1_w,   b_w1,   768 * 3072);
  cvt(w2_w,   b_w2,   3072 * 768);

  // LN1
  k_layernorm<<<ROWS / 8, 256, 0, stream>>>(x, ln1_g, ln1_b, b_ln1);

  // QKV = ln1 @ wqkv    [8192,768]x[768,2304] -> bf16
  k_gemm<2304, 768, false, true, false, false>
      <<<dim3(2304 / 128, ROWS / 128), 256, 0, stream>>>(
          b_ln1, b_wqkv, nullptr, nullptr, b_qkv);

  // attention -> bf16 [8192,768]
  k_attention<<<dim3(4096 / 64, 2 * 12), 128, 0, stream>>>(b_qkv, b_attn);

  // x1 = attn @ wo + wo_b + x   (fp32)
  k_gemm<768, 768, false, false, true, true>
      <<<dim3(768 / 128, ROWS / 128), 256, 0, stream>>>(
          b_attn, b_wo, wo_b, x, b_x1);

  // LN2
  k_layernorm<<<ROWS / 8, 256, 0, stream>>>(b_x1, ln2_g, ln2_b, b_ln2);

  // h1 = gelu(ln2 @ w1 + w1_b)  -> bf16 [8192,3072]
  k_gemm<3072, 768, true, true, true, false>
      <<<dim3(3072 / 128, ROWS / 128), 256, 0, stream>>>(
          b_ln2, b_w1, w1_b, nullptr, b_h1);

  // out = h1 @ w2 + w2_b + x1   (fp32 -> d_out)
  k_gemm<768, 3072, false, false, true, true>
      <<<dim3(768 / 128, ROWS / 128), 256, 0, stream>>>(
          b_h1, b_w2, w2_b, b_x1, (float*)d_out);
}
